// RNN_9775345566002
// MI455X (gfx1250) — matrix operand order repaired and checked by reading, not compiled, not run on hardware
//
#include <hip/hip_runtime.h>
#include <hip/hip_bf16.h>
#include <math.h>

// Problem constants
#define SEQ    2048
#define BATCH  64
#define INDIM  256
#define HID    256
#define LAYERS 2

typedef __attribute__((ext_vector_type(16))) __bf16        v16bf;
typedef __attribute__((ext_vector_type(8)))  float         v8f;
typedef __attribute__((ext_vector_type(4))) unsigned int   u32x4;
typedef __attribute__((ext_vector_type(8)))  int           i32x8;
typedef __attribute__((ext_vector_type(4)))  int           i32x4;

// LDS layout for the recurrence kernel (bytes)
#define U_LDS_OFF     0          // 128 KB bf16 U
#define H_LDS_OFF     131072     // 32 KB bf16 h
#define PRE_LDS_OFF   163840     // 2 x 64 KB f32 pre double-buffer
#define PRE_BUF_BYTES 65536
#define LDS2_BYTES    (PRE_LDS_OFF + 2 * PRE_BUF_BYTES)   // 294912

// ---- helpers ---------------------------------------------------------------

__device__ __forceinline__ unsigned short f2bf(float f) {
    unsigned int u = __builtin_bit_cast(unsigned int, f);
    unsigned int r = u + 0x7fffu + ((u >> 16) & 1u);   // round-to-nearest-even
    return (unsigned short)(r >> 16);
}

// A-matrix 16x32 bf16 fragment (M x K), rows row-major with `stride` halfs.
// lane L holds row (L&15); lanes<16 K {0..7,16..23}, lanes>=16 K {8..15,24..31}
__device__ __forceinline__ v16bf load_a_frag(const unsigned short* base, int stride,
                                             int row, int kb, int lane) {
    int off = (lane & 16) ? 8 : 0;
    const unsigned short* p = base + row * stride + kb + off;
    union { u32x4 q[2]; v16bf v; } f;
    f.q[0] = *(const u32x4*)(p);
    f.q[1] = *(const u32x4*)(p + 16);
    return f.v;
}

// B-matrix 32x16 bf16 fragment (K x N) where memory holds Bt[n, k] row-major
// (U[g,h] / W[h,i] exactly as given). lane L holds column (L&15);
// lanes<16 K {0..7,16..23}, lanes>=16 K {8..15,24..31}
__device__ __forceinline__ v16bf load_b_frag(const unsigned short* base, int stride,
                                             int col, int kb, int lane) {
    int off = (lane & 16) ? 8 : 0;
    const unsigned short* p = base + col * stride + kb + off;
    union { u32x4 q[2]; v16bf v; } f;
    f.q[0] = *(const u32x4*)(p);
    f.q[1] = *(const u32x4*)(p + 16);
    return f.v;
}

// ---- kernel 1: pre[l,t,b,h] = x[t,b,:] . W[l,h,:] + bias[l,h] --------------
// grid = (SEQ*BATCH/64, LAYERS), block = 256 (8 wave32)

extern "C" __global__ __launch_bounds__(256)
void rnn_precompute(const float* __restrict__ x, const float* __restrict__ W,
                    const float* __restrict__ bias, float* __restrict__ pre) {
    extern __shared__ char smem[];
    unsigned short* w_lds = (unsigned short*)smem;              // 256*256
    unsigned short* x_lds = (unsigned short*)(smem + 131072);   // 64*256
    float*          b_lds = (float*)(smem + 163840);            // 256

    const int tid = threadIdx.x;
    const int rb  = blockIdx.x;        // 64-row block of (t*BATCH+b)
    const int l   = blockIdx.y;

    const float* Wl = W + (size_t)l * HID * INDIM;
    for (int i = tid; i < HID * INDIM; i += 256) w_lds[i] = f2bf(Wl[i]);
    const float* xb = x + (size_t)rb * 64 * INDIM;
    for (int i = tid; i < 64 * INDIM; i += 256) x_lds[i] = f2bf(xb[i]);
    if (tid < HID) b_lds[tid] = bias[l * HID + tid];
    __syncthreads();

    const int lane   = tid & 31;
    const int wave   = tid >> 5;
    const int m_pair = wave & 1;
    const int n_grp  = wave >> 1;

    v8f acc[2][4];
    for (int mi = 0; mi < 2; ++mi)
        for (int n = 0; n < 4; ++n) acc[mi][n] = (v8f)(0.0f);

    for (int kt = 0; kt < INDIM / 32; ++kt) {
        const int kb = kt * 32;
        v16bf bfrag[4];
        for (int n = 0; n < 4; ++n) {
            int col = (n_grp * 4 + n) * 16 + (lane & 15);
            bfrag[n] = load_b_frag(w_lds, INDIM, col, kb, lane);
        }
        for (int mi = 0; mi < 2; ++mi) {
            int row = (m_pair * 2 + mi) * 16 + (lane & 15);
            v16bf afrag = load_a_frag(x_lds, INDIM, row, kb, lane);
            for (int n = 0; n < 4; ++n)
                acc[mi][n] = __builtin_amdgcn_wmma_f32_16x16x32_bf16(
                    false, afrag, false, bfrag[n], (short)0, acc[mi][n], false, false);
        }
    }

    const int rhalf = (lane & 16) ? 8 : 0;
    for (int mi = 0; mi < 2; ++mi) {
        for (int n = 0; n < 4; ++n) {
            int col = (n_grp * 4 + n) * 16 + (lane & 15);
            float bv = b_lds[col];
            for (int r = 0; r < 8; ++r) {
                int m = (m_pair * 2 + mi) * 16 + rhalf + r;
                size_t idx = (((size_t)l * SEQ * BATCH) + (size_t)rb * 64 + m) * HID + col;
                __builtin_nontemporal_store(acc[mi][n][r] + bv, pre + idx);
            }
        }
    }
}

// ---- TDM issue: load one 64KB pre step block into an LDS buffer ------------

__device__ __forceinline__ void tdm_load_pre(const float* gptr, unsigned lds_off) {
    unsigned long long ga = (unsigned long long)(uintptr_t)gptr;
    u32x4 g0;
    g0[0] = 1u;                                   // count=1, no gather
    g0[1] = lds_off;                              // lds_addr
    g0[2] = (unsigned)(ga & 0xffffffffu);         // global_addr[31:0]
    g0[3] = (unsigned)(ga >> 32) | 0x80000000u;   // global_addr[56:32] | type=2
    i32x8 g1;
    g1[0] = 0x00020000;   // workgroup_mask=0, data_size=2 (4 bytes)
    g1[1] = 0x40000000;   // tensor_dim0[15:0]=16384 in [31:16]
    g1[2] = 0x00010000;   // tensor_dim0[31:16]=0, tensor_dim1[15:0]=1
    g1[3] = 0x40000000;   // tile_dim0=16384 in [31:16]
    g1[4] = 0x00000001;   // tile_dim1=1, tile_dim2=0
    g1[5] = 16384;        // tensor_dim0_stride[31:0]
    g1[6] = 0;
    g1[7] = 0;
    i32x4 z4 = (i32x4)(0);
    i32x8 z8 = (i32x8)(0);
    // amdgpu-toolchain (clang-23) 6-arg form
    __builtin_amdgcn_tensor_load_to_lds(g0, g1, z4, z4, z8, 0);
}

// ---- kernel 2: sequential recurrence, one workgroup per layer --------------
// h[l] = tanh(pre[l,t] + h[l] . U[l]^T); output = h of last layer each step.
// LDS: U bf16 128KB | h bf16 32KB | pre f32 2x64KB (TDM double buffer)

extern "C" __global__ __launch_bounds__(256)
void rnn_recurrence(const float* __restrict__ h0, const float* __restrict__ U,
                    const float* __restrict__ pre, float* __restrict__ out_seq,
                    float* __restrict__ out_hn) {
    extern __shared__ char smem[];
    unsigned short* u_lds = (unsigned short*)(smem + U_LDS_OFF);   // 256*256
    unsigned short* h_lds = (unsigned short*)(smem + H_LDS_OFF);   // 64*256

    const int tid = threadIdx.x;
    const int l   = blockIdx.x;
    const float* pre_l = pre + (size_t)l * SEQ * BATCH * HID;

    // wave 0 kicks off the TDM fetch of pre[0] before staging
    if ((tid >> 5) == 0) {
        tdm_load_pre(pre_l, PRE_LDS_OFF);
    }

    const float* Ul = U + (size_t)l * HID * HID;
    for (int i = tid; i < HID * HID; i += 256) u_lds[i] = f2bf(Ul[i]);
    const float* h0l = h0 + (size_t)l * BATCH * HID;
    for (int i = tid; i < BATCH * HID; i += 256) h_lds[i] = f2bf(h0l[i]);
    __syncthreads();

    const int lane   = tid & 31;
    const int wave   = tid >> 5;
    const int m_pair = wave & 1;
    const int n_grp  = wave >> 1;
    const int rhalf  = (lane & 16) ? 8 : 0;
    const bool last_layer = (l == LAYERS - 1);

    for (int t = 0; t < SEQ; ++t) {
        // wave 0 queues pre[t+1] into the other buffer while we compute
        if (wave == 0) {
            if (t + 1 < SEQ)
                tdm_load_pre(pre_l + (size_t)(t + 1) * BATCH * HID,
                             PRE_LDS_OFF + (unsigned)((t + 1) & 1) * PRE_BUF_BYTES);
        }

        v8f acc[2][4];
        for (int mi = 0; mi < 2; ++mi)
            for (int n = 0; n < 4; ++n) acc[mi][n] = (v8f)(0.0f);

        for (int kt = 0; kt < HID / 32; ++kt) {
            const int kb = kt * 32;
            v16bf bfrag[4];
            for (int n = 0; n < 4; ++n) {
                int col = (n_grp * 4 + n) * 16 + (lane & 15);
                bfrag[n] = load_b_frag(u_lds, HID, col, kb, lane);
            }
            for (int mi = 0; mi < 2; ++mi) {
                int row = (m_pair * 2 + mi) * 16 + (lane & 15);
                v16bf afrag = load_a_frag(h_lds, HID, row, kb, lane);
                for (int n = 0; n < 4; ++n)
                    acc[mi][n] = __builtin_amdgcn_wmma_f32_16x16x32_bf16(
                        false, afrag, false, bfrag[n], (short)0, acc[mi][n], false, false);
            }
        }

        // wave 0 guarantees pre[t] has landed: with t+1 queued we have at most
        // 2 outstanding TDM ops (wait <=1); on the last step only pre[t] is
        // outstanding (wait <=0).
        if (wave == 0) {
            if (t + 1 < SEQ) __builtin_amdgcn_s_wait_tensorcnt(1);
            else             __builtin_amdgcn_s_wait_tensorcnt(0);
        }
        __syncthreads();   // h reads done for step t + pre[t] visible to all

        const float* pf = (const float*)(smem + PRE_LDS_OFF
                                         + (size_t)(t & 1) * PRE_BUF_BYTES);
        for (int mi = 0; mi < 2; ++mi) {
            for (int n = 0; n < 4; ++n) {
                int col = (n_grp * 4 + n) * 16 + (lane & 15);
                for (int r = 0; r < 8; ++r) {
                    int m = (m_pair * 2 + mi) * 16 + rhalf + r;
                    float v  = acc[mi][n][r] + pf[m * HID + col];
                    float th = tanhf(v);
                    h_lds[m * HID + col] = f2bf(th);
                    if (last_layer)
                        __builtin_nontemporal_store(
                            th, out_seq + (size_t)t * BATCH * HID + m * HID + col);
                    if (t == SEQ - 1)
                        out_hn[((size_t)l * BATCH + m) * HID + col] = th;
                }
            }
        }

        __syncthreads();   // new h visible before step t+1 reads
    }
}

// ---- launcher --------------------------------------------------------------

extern "C" void kernel_launch(void* const* d_in, const int* in_sizes, int n_in,
                              void* d_out, int out_size, void* d_ws, size_t ws_size,
                              hipStream_t stream) {
    const float* x    = (const float*)d_in[0];
    const float* h_0  = (const float*)d_in[1];
    const float* W    = (const float*)d_in[2];
    const float* U    = (const float*)d_in[3];
    const float* bias = (const float*)d_in[4];

    float* out_seq = (float*)d_out;                        // [SEQ,B,H]
    float* out_hn  = out_seq + (size_t)SEQ * BATCH * HID;  // [L,B,H]
    float* pre     = (float*)d_ws;                         // [L,SEQ,B,H]

    const size_t lds1 = 131072 + 32768 + 1024;   // W + x-block + bias
    const size_t lds2 = LDS2_BYTES;              // U + h + pre double buffer

    (void)hipFuncSetAttribute((const void*)rnn_precompute,
                              hipFuncAttributeMaxDynamicSharedMemorySize, (int)lds1);
    (void)hipFuncSetAttribute((const void*)rnn_recurrence,
                              hipFuncAttributeMaxDynamicSharedMemorySize, (int)lds2);

    dim3 grid1(SEQ * BATCH / 64, LAYERS);
    rnn_precompute<<<grid1, 256, lds1, stream>>>(x, W, bias, pre);

    dim3 grid2(LAYERS);
    rnn_recurrence<<<grid2, 256, lds2, stream>>>(h_0, U, pre, out_seq, out_hn);
}
